// Block1_3_77893526880843
// MI455X (gfx1250) — matrix operand order repaired and checked by reading, not compiled, not run on hardware
//
#include <hip/hip_runtime.h>

typedef __attribute__((ext_vector_type(16))) _Float16 v16h;
typedef __attribute__((ext_vector_type(8)))  float    v8f;

#define NB 8  // batch size

// =====================================================================
// Weight prep: fold conv bias + BN into per-channel (scale, shift),
// convert weights to f16 in GEMM layout [cout][Kpad] with
//   K = tap*CIN + ci   (CIN >= 32, Kpad = 9*CIN exactly)
//   K = ci*9 + tap     (CIN  < 32, Kpad = 32, zero padded)
// =====================================================================
__global__ void prep_weights_k(const float* __restrict__ w, const float* __restrict__ bias,
                               const float* __restrict__ gamma, const float* __restrict__ beta,
                               const float* __restrict__ rm, const float* __restrict__ rv,
                               _Float16* __restrict__ w16, float* __restrict__ ab,
                               int cout, int cin, int Kpad)
{
  int idx = blockIdx.x * blockDim.x + threadIdx.x;
  if (idx < cout) {
    float inv = gamma[idx] * rsqrtf(rv[idx] + 1e-5f);
    ab[idx] = inv;                                      // scale
    ab[cout + idx] = beta[idx] - inv * rm[idx] + bias[idx] * inv;  // shift (bias folded)
  }
  if (idx >= cout * Kpad) return;
  int n = idx / Kpad;
  int k = idx - n * Kpad;
  int ci, t;
  if (cin >= 32) { t = k / cin; ci = k - t * cin; }
  else           { ci = k / 9;  t  = k - ci * 9;  }
  float v = (ci < cin && t < 9) ? w[(n * cin + ci) * 9 + t] : 0.0f;
  w16[idx] = (_Float16)v;
}

// Async DMA: one dword global -> LDS, tracked by ASYNCcnt (CDNA5).
__device__ __forceinline__ void async_g2l_b32(unsigned lds_off, const float* gaddr)
{
  asm volatile("global_load_async_to_lds_b32 %0, %1, off"
               :: "v"(lds_off), "v"(gaddr) : "memory");
}

// =====================================================================
// conv3x3 (stride 1, pad=dilation=DIR) + BN + ReLU, implicit GEMM on
// v_wmma_f32_16x16x32_f16. Block = 128 threads = 4 waves:
//   wave tile = 16 flattened pixels (M) x 16 output channels (N)
//   block tile = 32 pixels x 32 channels, K consumed 32 at a time.
// A tile is double-buffer streamed into LDS with async global->LDS DMA
// (clamped addresses; OOB taps zeroed at fragment build via predicate).
// Input/output NCHW fp32 with channel-stride/offset (in-place concat).
// =====================================================================
template<int CIN, int DIR>
__global__ __launch_bounds__(128) void conv3x3_wmma_k(
    const float* __restrict__ in, int inCtot, int inCoff,
    const _Float16* __restrict__ w16, const float* __restrict__ ab,
    float* __restrict__ out, int outCtot, int outCoff,
    int H, int W, int coutTot)
{
  constexpr int KPAD = (CIN >= 32) ? CIN * 9 : 32;
  constexpr int SROW = 36;              // LDS row stride (dwords): 16B-aligned, bank-spread
  const int HW   = H * W;
  const int bb   = blockIdx.y;          // batch
  const int ct   = blockIdx.z;          // cout tile of 32
  const int lane = threadIdx.x & 31;
  const int wave = threadIdx.x >> 5;
  const int wm   = wave >> 1;           // M subtile (0/1)
  const int wn   = wave & 1;            // N subtile (0/1)
  const int n0   = ct * 32 + wn * 16;
  const int nIdx = n0 + (lane & 15);
  const int hiHalf = lane >> 4;         // ISA A/B lane-half selector
  const int p    = blockIdx.x * 32 + wm * 16 + (lane & 15);  // flat out pixel
  const bool pvalid = p < HW;
  const int y = pvalid ? (p / W) : 0;
  const int x = pvalid ? (p - y * W) : 0;
  const int bbase = (bb * inCtot + inCoff) * HW;
  const _Float16* wrow = w16 + nIdx * KPAD + hiHalf * 16;   // aligned v16h rows

  v8f acc = {0.f,0.f,0.f,0.f,0.f,0.f,0.f,0.f};

  if constexpr (CIN >= 32) {
    __shared__ float s_a[2][32 * SROW];
    const unsigned ldsb0 = (unsigned)(uintptr_t)&s_a[0][0];
    const unsigned ldsb1 = (unsigned)(uintptr_t)&s_a[1][0];

    // staging role: this thread streams channel `cidx` of pixels wid*8..wid*8+7
    const int cidx = threadIdx.x & 31;
    const int wid  = threadIdx.x >> 5;
    int sy[8], sx[8];
    #pragma unroll
    for (int l = 0; l < 8; ++l) {
      int pl = blockIdx.x * 32 + wid * 8 + l;
      bool v = pl < HW;
      int py = v ? pl / W : 0;
      sy[l] = py;
      sx[l] = v ? pl - py * W : 0;
    }

    auto stage = [&](unsigned ldsb, int kc) {
      const int t = kc / CIN;
      const int ciBase = kc - t * CIN;
      const int ky = t / 3, kx = t - ky * 3;
      const float* cb = in + bbase + (ciBase + cidx) * HW;
      #pragma unroll
      for (int l = 0; l < 8; ++l) {
        int yy = sy[l] + (ky - 1) * DIR; yy = yy < 0 ? 0 : (yy >= H ? H - 1 : yy);
        int xx = sx[l] + (kx - 1) * DIR; xx = xx < 0 ? 0 : (xx >= W ? W - 1 : xx);
        unsigned lo = ldsb + (unsigned)(((wid * 8 + l) * SROW + cidx) * 4);
        async_g2l_b32(lo, cb + yy * W + xx);   // 8 ASYNCcnt ops per wave per chunk
      }
    };

    stage(ldsb0, 0);
    int bi = 0;

    #pragma unroll 1
    for (int kc = 0; kc < KPAD; kc += 32) {
      if (kc + 32 < KPAD) {
        __builtin_prefetch(wrow + kc + 32, 0, 1);     // global_prefetch_b8
        stage(bi ? ldsb0 : ldsb1, kc + 32);
        asm volatile("s_wait_asynccnt 0x8" ::: "memory");  // chunk kc landed
      } else {
        asm volatile("s_wait_asynccnt 0x0" ::: "memory");
      }
      __syncthreads();   // cross-wave: all 32 pixels staged

      // per-lane zero predicate for this tap (uniform branch-free)
      const int t  = kc / CIN;
      const int ky = t / 3, kx = t - ky * 3;
      const int yy = y + (ky - 1) * DIR;
      const int xx = x + (kx - 1) * DIR;
      const float zsel = (pvalid && yy >= 0 && yy < H && xx >= 0 && xx < W) ? 1.0f : 0.0f;

      // A fragment in contiguous k order (k = 16*hiHalf + s, same as B), from LDS (4x ds_load_b128)
      const int srow = (wm * 16 + (lane & 15)) * SROW + hiHalf * 16;
      float4 f0 = *(const float4*)&s_a[bi][srow];
      float4 f1 = *(const float4*)&s_a[bi][srow + 4];
      float4 f2 = *(const float4*)&s_a[bi][srow + 8];
      float4 f3 = *(const float4*)&s_a[bi][srow + 12];
      v16h a;
      a[0]=(_Float16)(f0.x*zsel); a[1]=(_Float16)(f0.y*zsel);
      a[2]=(_Float16)(f0.z*zsel); a[3]=(_Float16)(f0.w*zsel);
      a[4]=(_Float16)(f1.x*zsel); a[5]=(_Float16)(f1.y*zsel);
      a[6]=(_Float16)(f1.z*zsel); a[7]=(_Float16)(f1.w*zsel);
      a[8]=(_Float16)(f2.x*zsel); a[9]=(_Float16)(f2.y*zsel);
      a[10]=(_Float16)(f2.z*zsel); a[11]=(_Float16)(f2.w*zsel);
      a[12]=(_Float16)(f3.x*zsel); a[13]=(_Float16)(f3.y*zsel);
      a[14]=(_Float16)(f3.z*zsel); a[15]=(_Float16)(f3.w*zsel);

      // B fragment: single aligned 32B load (lanes0-15: K kc..kc+15; lanes16-31: +16)
      v16h bf = *(const v16h*)(wrow + kc);
      acc = __builtin_amdgcn_wmma_f32_16x16x32_f16(false, a, false, bf,
                                                   (short)0, acc, false, false);
      __syncthreads();   // everyone done reading before buffer is restaged
      bi ^= 1;
    }
  } else {
    // CIN == 3: single chunk, K = ci*9 + tap, padded with zeros to 32
    v16h a;
    #pragma unroll
    for (int h = 0; h < 16; ++h) {
      const int base = h;
      const int K  = base + hiHalf * 16;
      const int ci = (K * 57) >> 9;       // K/9 for K<32
      const int t  = K - ci * 9;
      const int ky = (t * 11) >> 5;       // t/3 for t<9
      const int kx = t - ky * 3;
      const int yy = y + (ky - 1) * DIR;
      const int xx = x + (kx - 1) * DIR;
      const bool inb = pvalid && (K < CIN * 9) &&
                       (yy >= 0) && (yy < H) && (xx >= 0) && (xx < W);
      a[h] = (_Float16)(inb ? in[bbase + ci * HW + yy * W + xx] : 0.0f);
    }
    v16h bf = *(const v16h*)wrow;
    acc = __builtin_amdgcn_wmma_f32_16x16x32_f16(false, a, false, bf,
                                                 (short)0, acc, false, false);
  }

  // Fused BN + ReLU epilogue; D layout: lane<16 -> M=r, lane>=16 -> M=8+r
  const float scale = ab[nIdx];
  const float shift = ab[coutTot + nIdx];
  const int mBase = blockIdx.x * 32 + wm * 16 + hiHalf * 8;
  const int obase = (bb * outCtot + outCoff + nIdx) * HW;
  float vals[8];
  #pragma unroll
  for (int r = 0; r < 8; ++r) {
    float v = acc[r] * scale + shift;
    vals[r] = v > 0.0f ? v : 0.0f;
  }
  if ((mBase + 7 < HW) && ((((unsigned)(obase + mBase)) & 3u) == 0u)) {
    float4* o = (float4*)(out + obase + mBase);
    o[0] = make_float4(vals[0], vals[1], vals[2], vals[3]);
    o[1] = make_float4(vals[4], vals[5], vals[6], vals[7]);
  } else {
    #pragma unroll
    for (int r = 0; r < 8; ++r)
      if (mBase + r < HW) out[obase + mBase + r] = vals[r];
  }
}

// =====================================================================
// 2x2 maxpool, stride 2 (input may live inside a concat buffer)
// =====================================================================
__global__ void maxpool2_k(const float* __restrict__ in, int inCt, int inCo,
                           float* __restrict__ out, int C, int ho, int wo, int n)
{
  int idx = blockIdx.x * blockDim.x + threadIdx.x;
  if (idx >= n) return;
  int hw = ho * wo;
  int b  = idx / (C * hw);
  int r  = idx - b * C * hw;
  int c  = r / hw;
  int rr = r - c * hw;
  int i  = rr / wo;
  int j  = rr - i * wo;
  int wi = wo * 2;
  const float* p = in + ((size_t)(b * inCt + inCo + c) * (ho * 2) + 2 * i) * wi + 2 * j;
  out[idx] = fmaxf(fmaxf(p[0], p[1]), fmaxf(p[wi], p[wi + 1]));
}

// =====================================================================
// Bilinear resize (jax.image.resize, antialias=False): src=(i+.5)*s-.5
// =====================================================================
__global__ void bilinear_k(const float* __restrict__ in, float* __restrict__ out,
                           int C, int hi, int wi, int ho, int wo, int n)
{
  int idx = blockIdx.x * blockDim.x + threadIdx.x;
  if (idx >= n) return;
  int hw = ho * wo;
  int b  = idx / (C * hw);
  int r  = idx - b * C * hw;
  int c  = r / hw;
  int rr = r - c * hw;
  int i  = rr / wo;
  int j  = rr - i * wo;
  float sy = (i + 0.5f) * (float)hi / (float)ho - 0.5f;
  float sx = (j + 0.5f) * (float)wi / (float)wo - 0.5f;
  float fy0 = floorf(sy), fx0 = floorf(sx);
  int i0 = (int)fy0, j0 = (int)fx0;
  float fy = sy - fy0, fx = sx - fx0;
  int i0c = i0 < 0 ? 0 : (i0 >= hi ? hi - 1 : i0);
  int i1c = (i0 + 1) < 0 ? 0 : ((i0 + 1) >= hi ? hi - 1 : i0 + 1);
  int j0c = j0 < 0 ? 0 : (j0 >= wi ? wi - 1 : j0);
  int j1c = (j0 + 1) < 0 ? 0 : ((j0 + 1) >= wi ? wi - 1 : j0 + 1);
  const float* p = in + (size_t)(b * C + c) * hi * wi;
  float v00 = p[i0c * wi + j0c], v01 = p[i0c * wi + j1c];
  float v10 = p[i1c * wi + j0c], v11 = p[i1c * wi + j1c];
  out[idx] = (1.f - fy) * ((1.f - fx) * v00 + fx * v01)
           +        fy  * ((1.f - fx) * v10 + fx * v11);
}

// =====================================================================
// Guided filter (r=2, eps=.01). The cumsum/diff box filter equals a
// truncated 5x5 window sum; count = clipped window area.
// =====================================================================
__global__ void gf_guide_k(const float* __restrict__ g, float* __restrict__ stats,
                           int h, int w, int ng)
{
  int idx = blockIdx.x * blockDim.x + threadIdx.x;
  if (idx >= ng) return;
  int hw = h * w;
  int b = idx / hw;
  int r = idx - b * hw;
  int i = r / w;
  int j = r - i * w;
  int i0 = i - 2 > 0 ? i - 2 : 0, i1 = i + 2 < h - 1 ? i + 2 : h - 1;
  int j0 = j - 2 > 0 ? j - 2 : 0, j1 = j + 2 < w - 1 ? j + 2 : w - 1;
  float sx = 0.f, sxx = 0.f;
  const float* p = g + (size_t)b * hw;
  for (int ii = i0; ii <= i1; ++ii)
    for (int jj = j0; jj <= j1; ++jj) { float v = p[ii * w + jj]; sx += v; sxx += v * v; }
  float cnt = (float)((i1 - i0 + 1) * (j1 - j0 + 1));
  float mx = sx / cnt;
  stats[idx]      = mx;                   // mean_x
  stats[ng + idx] = sxx / cnt - mx * mx;  // var_x
}

__global__ void gf_ab_k(const float* __restrict__ g, const float* __restrict__ y,
                        const float* __restrict__ stats,
                        float* __restrict__ A, float* __restrict__ Bb,
                        int C, int h, int w, int n, int ng)
{
  int idx = blockIdx.x * blockDim.x + threadIdx.x;
  if (idx >= n) return;
  int hw = h * w;
  int b  = idx / (C * hw);
  int r  = idx - b * C * hw;
  int c  = r / hw;
  int rr = r - c * hw;
  int i  = rr / w;
  int j  = rr - i * w;
  int i0 = i - 2 > 0 ? i - 2 : 0, i1 = i + 2 < h - 1 ? i + 2 : h - 1;
  int j0 = j - 2 > 0 ? j - 2 : 0, j1 = j + 2 < w - 1 ? j + 2 : w - 1;
  const float* pg = g + (size_t)b * hw;
  const float* py = y + (size_t)(b * C + c) * hw;
  float sy = 0.f, sxy = 0.f;
  for (int ii = i0; ii <= i1; ++ii)
    for (int jj = j0; jj <= j1; ++jj) {
      float xv = pg[ii * w + jj];
      float yv = py[ii * w + jj];
      sy += yv; sxy += xv * yv;
    }
  float cnt = (float)((i1 - i0 + 1) * (j1 - j0 + 1));
  int gi = b * hw + i * w + j;
  float mx = stats[gi], vx = stats[ng + gi];
  float my = sy / cnt;
  float cov = sxy / cnt - mx * my;
  float a = cov / (vx + 0.01f);
  A[idx]  = a;
  Bb[idx] = my - a * mx;
}

__global__ void gf_out_k(const float* __restrict__ g,
                         const float* __restrict__ A, const float* __restrict__ Bb,
                         float* __restrict__ out, int outCt, int outCo,
                         int C, int h, int w, int n)
{
  int idx = blockIdx.x * blockDim.x + threadIdx.x;
  if (idx >= n) return;
  int hw = h * w;
  int b  = idx / (C * hw);
  int r  = idx - b * C * hw;
  int c  = r / hw;
  int rr = r - c * hw;
  int i  = rr / w;
  int j  = rr - i * w;
  int i0 = i - 2 > 0 ? i - 2 : 0, i1 = i + 2 < h - 1 ? i + 2 : h - 1;
  int j0 = j - 2 > 0 ? j - 2 : 0, j1 = j + 2 < w - 1 ? j + 2 : w - 1;
  const float* pa = A  + (size_t)(b * C + c) * hw;
  const float* pb = Bb + (size_t)(b * C + c) * hw;
  float sa = 0.f, sb = 0.f;
  for (int ii = i0; ii <= i1; ++ii)
    for (int jj = j0; jj <= j1; ++jj) { sa += pa[ii * w + jj]; sb += pb[ii * w + jj]; }
  float cnt = (float)((i1 - i0 + 1) * (j1 - j0 + 1));
  float res = (sa / cnt) * g[(size_t)b * hw + i * w + j] + sb / cnt;
  out[(size_t)(b * outCt + outCo + c) * hw + i * w + j] = res;
}

__global__ void add_k(float* __restrict__ out, const float* __restrict__ a, int n)
{
  int idx = blockIdx.x * blockDim.x + threadIdx.x;
  if (idx < n) out[idx] += a[idx];
}

// =====================================================================
// Host orchestration
// =====================================================================
struct LSpec { int cin, cout; };
static const LSpec SPECS[14] = {
  {3,64},{64,32},{32,32},{32,32},{32,32},{32,32},{32,32},{32,32},
  {64,32},{64,32},{64,32},{64,32},{64,32},{64,64}
};

extern "C" void kernel_launch(void* const* d_in, const int* in_sizes, int n_in,
                              void* d_out, int out_size, void* d_ws, size_t ws_size,
                              hipStream_t stream)
{
  (void)in_sizes; (void)n_in; (void)out_size; (void)ws_size;
  const float* x   = (const float*)d_in[0];
  const float* img = (const float*)d_in[1];
  float* const o   = (float*)d_out;

  // output regions (hx6d, hx5d, hx4d, hx3d, hx2d, hx1d+hxin)
  float* r0 = o;                 // 8*32*9*9
  float* r1 = o + 20736;         // 8*32*18*18
  float* r2 = o + 103680;        // 8*32*36*36
  float* r3 = o + 435456;        // 8*32*72*72
  float* r4 = o + 1762560;       // 8*32*144*144
  float* r5 = o + 7070976;       // 8*64*288*288

  // bump allocator over workspace
  size_t off = 0;
  char* wsb = (char*)d_ws;
  auto alloc = [&](size_t bytes) -> void* {
    void* p = wsb + off;
    off += (bytes + 255) & ~(size_t)255;
    return p;
  };

  // ---- per-layer f16 weights + folded BN scale/shift ----
  _Float16* w16[14]; float* ab[14];
  for (int i = 0; i < 14; ++i) {
    int cin = SPECS[i].cin, cout = SPECS[i].cout;
    int Kpad = (cin >= 32) ? cin * 9 : 32;
    w16[i] = (_Float16*)alloc((size_t)cout * Kpad * sizeof(_Float16));
    ab[i]  = (float*)alloc((size_t)2 * cout * sizeof(float));
    const float* pw  = (const float*)d_in[2 + 6*i + 0];
    const float* pb  = (const float*)d_in[2 + 6*i + 1];
    const float* pg  = (const float*)d_in[2 + 6*i + 2];
    const float* pbe = (const float*)d_in[2 + 6*i + 3];
    const float* prm = (const float*)d_in[2 + 6*i + 4];
    const float* prv = (const float*)d_in[2 + 6*i + 5];
    int tot = cout * Kpad;
    prep_weights_k<<<(tot + 255)/256, 256, 0, stream>>>(
        pw, pb, pg, pbe, prm, prv, w16[i], ab[i], cout, cin, Kpad);
  }

  // ---- activation / scratch tensors ----
  float* hxin = (float*)alloc((size_t)NB*64*288*288*4);
  float* cat1 = (float*)alloc((size_t)NB*64*288*288*4);
  float* cat2 = (float*)alloc((size_t)NB*64*144*144*4);
  float* cat3 = (float*)alloc((size_t)NB*64*72*72*4);
  float* cat4 = (float*)alloc((size_t)NB*64*36*36*4);
  float* cat5 = (float*)alloc((size_t)NB*64*18*18*4);
  float* cat6 = (float*)alloc((size_t)NB*64*9*9*4);
  float* im2  = (float*)alloc((size_t)NB*144*144*4);
  float* im3  = (float*)alloc((size_t)NB*72*72*4);
  float* im4  = (float*)alloc((size_t)NB*36*36*4);
  float* im5  = (float*)alloc((size_t)NB*18*18*4);
  float* im6  = (float*)alloc((size_t)NB*9*9*4);
  float* pool = (float*)alloc((size_t)NB*32*144*144*4);   // reused per level
  float* hxh  = (float*)alloc((size_t)NB*32*144*144*4);   // encoder GF output
  float* stats= (float*)alloc((size_t)2*NB*288*288*4);    // GF guide mean/var
  float* Abuf = (float*)alloc((size_t)NB*32*288*288*4);
  float* Bbuf = (float*)alloc((size_t)NB*32*288*288*4);
  float* ubuf = (float*)alloc((size_t)NB*32*288*288*4);   // decoder upsample

  auto conv = [&](const float* in, int inCt, int inCo, int li,
                  float* outp, int outCt, int outCo, int H, int W, int dir) {
    int cout = SPECS[li].cout, cin = SPECS[li].cin;
    dim3 g((H*W + 31)/32, NB, cout/32);
    dim3 b(128, 1, 1);
    if (cin == 3)
      conv3x3_wmma_k<3,1><<<g,b,0,stream>>>(in,inCt,inCo,w16[li],ab[li],outp,outCt,outCo,H,W,cout);
    else if (cin == 32 && dir == 1)
      conv3x3_wmma_k<32,1><<<g,b,0,stream>>>(in,inCt,inCo,w16[li],ab[li],outp,outCt,outCo,H,W,cout);
    else if (cin == 32 && dir == 2)
      conv3x3_wmma_k<32,2><<<g,b,0,stream>>>(in,inCt,inCo,w16[li],ab[li],outp,outCt,outCo,H,W,cout);
    else
      conv3x3_wmma_k<64,1><<<g,b,0,stream>>>(in,inCt,inCo,w16[li],ab[li],outp,outCt,outCo,H,W,cout);
  };

  auto gfilter = [&](const float* g, const float* yv, int C, int h, int w,
                     float* outp, int outCt, int outCo) {
    int ng = NB * h * w;
    gf_guide_k<<<(ng + 255)/256, 256, 0, stream>>>(g, stats, h, w, ng);
    int n = NB * C * h * w;
    gf_ab_k<<<(n + 255)/256, 256, 0, stream>>>(g, yv, stats, Abuf, Bbuf, C, h, w, n, ng);
    gf_out_k<<<(n + 255)/256, 256, 0, stream>>>(g, Abuf, Bbuf, outp, outCt, outCo, C, h, w, n);
  };

  auto mpool = [&](const float* in, int inCt, int inCo, float* outp, int C, int ho, int wo) {
    int n = NB * C * ho * wo;
    maxpool2_k<<<(n + 255)/256, 256, 0, stream>>>(in, inCt, inCo, outp, C, ho, wo, n);
  };

  auto resize = [&](const float* in, float* outp, int C, int hi, int wi, int ho, int wo) {
    int n = NB * C * ho * wo;
    bilinear_k<<<(n + 255)/256, 256, 0, stream>>>(in, outp, C, hi, wi, ho, wo, n);
  };

  // ================= Encoder =================
  conv(x,    3,  0, 0,  hxin, 64, 0, 288, 288, 1);   // convin -> hxin (64ch)
  conv(hxin, 64, 0, 1,  cat1, 64, 0, 288, 288, 1);   // rbc1 -> hx1 = cat1[:, :32]

  resize(img, im2, 1, 288, 288, 144, 144);
  mpool(cat1, 64, 0, pool, 32, 144, 144);
  gfilter(im2, pool, 32, 144, 144, hxh, 32, 0);      // hx1h
  conv(hxh, 32, 0, 2, cat2, 64, 0, 144, 144, 1);     // rbc2 -> hx2

  resize(im2, im3, 1, 144, 144, 72, 72);
  mpool(cat2, 64, 0, pool, 32, 72, 72);
  gfilter(im3, pool, 32, 72, 72, hxh, 32, 0);        // hx2h
  conv(hxh, 32, 0, 3, cat3, 64, 0, 72, 72, 1);       // rbc3 -> hx3

  resize(im3, im4, 1, 72, 72, 36, 36);
  mpool(cat3, 64, 0, pool, 32, 36, 36);
  gfilter(im4, pool, 32, 36, 36, hxh, 32, 0);        // hx3h
  conv(hxh, 32, 0, 4, cat4, 64, 0, 36, 36, 1);       // rbc4 -> hx4

  resize(im4, im5, 1, 36, 36, 18, 18);
  mpool(cat4, 64, 0, pool, 32, 18, 18);
  gfilter(im5, pool, 32, 18, 18, hxh, 32, 0);        // hx4h
  conv(hxh, 32, 0, 5, cat5, 64, 0, 18, 18, 1);       // rbc5 -> hx5

  resize(im5, im6, 1, 18, 18, 9, 9);
  mpool(cat5, 64, 0, pool, 32, 9, 9);
  gfilter(im6, pool, 32, 9, 9, hxh, 32, 0);          // hx5h
  conv(hxh, 32, 0, 6, cat6, 64, 0, 9, 9, 1);         // rbc6 -> hx6 = cat6[:, :32]

  conv(cat6, 64, 0, 7, cat6, 64, 32, 9, 9, 2);       // rbc7 (dilation 2) -> hx7 = cat6[:, 32:]
  conv(cat6, 64, 0, 8, r0,   32, 0, 9, 9, 1);        // rbc6d -> hx6d (output 0)

  // ================= Decoder =================
  resize(r0, ubuf, 32, 9, 9, 18, 18);
  gfilter(im5, ubuf, 32, 18, 18, cat5, 64, 32);      // h -> cat5[:, 32:]
  conv(cat5, 64, 0, 9, r1, 32, 0, 18, 18, 1);        // rbc5d -> hx5d

  resize(r1, ubuf, 32, 18, 18, 36, 36);
  gfilter(im4, ubuf, 32, 36, 36, cat4, 64, 32);
  conv(cat4, 64, 0, 10, r2, 32, 0, 36, 36, 1);       // rbc4d -> hx4d

  resize(r2, ubuf, 32, 36, 36, 72, 72);
  gfilter(im3, ubuf, 32, 72, 72, cat3, 64, 32);
  conv(cat3, 64, 0, 11, r3, 32, 0, 72, 72, 1);       // rbc3d -> hx3d

  resize(r3, ubuf, 32, 72, 72, 144, 144);
  gfilter(im2, ubuf, 32, 144, 144, cat2, 64, 32);
  conv(cat2, 64, 0, 12, r4, 32, 0, 144, 144, 1);     // rbc2d -> hx2d

  resize(r4, ubuf, 32, 144, 144, 288, 288);
  gfilter(img, ubuf, 32, 288, 288, cat1, 64, 32);
  conv(cat1, 64, 0, 13, r5, 64, 0, 288, 288, 1);     // rbc1d -> hx1d

  int nres = NB * 64 * 288 * 288;
  add_k<<<(nres + 255)/256, 256, 0, stream>>>(r5, hxin, nres);  // + hxin
}
